// Model_22076131901860
// MI455X (gfx1250) — matrix operand order repaired and checked by reading, not compiled, not run on hardware
//
#include <hip/hip_runtime.h>

// ---------------------------------------------------------------------------
// Fused 2-layer recurrent LIF SNN + readout, persistent kernel for gfx1250.
// One workgroup (7 wave32s, 224 threads) owns a 16-row batch tile and scans
// all T=1000 timesteps. Hot weights (Win1/Wrec1/Win2) live in registers as
// WMMA B-fragments; Wrec2/Wout fragments are pre-packed into LDS in lane
// order (two ds_load_b128 per use). Spikes ping-pong through LDS. All GEMMs
// are v_wmma_f32_16x16x32_f16. Readout for step t-1 is overlapped with
// phase A of step t, so only two barriers per timestep.
// ---------------------------------------------------------------------------

typedef __attribute__((ext_vector_type(16))) _Float16 v16h;
typedef __attribute__((ext_vector_type(8)))  _Float16 v8h;
typedef __attribute__((ext_vector_type(4)))  _Float16 v4h;
typedef __attribute__((ext_vector_type(4)))  float    v4f;
typedef __attribute__((ext_vector_type(8)))  float    v8f;

#define B_TOT    256
#define T_LEN    1000
#define D_IN     128
#define H_N      100
#define O_N      10
#define MT       16      // batch rows per workgroup (WMMA M)
#define KP       128     // padded K (D=128 exactly; H=100 padded to 128)
#define NWAVES   7       // N tiles: 7*16 = 112 >= 100
#define NTHREADS (NWAVES * 32)
#define THRESH   1.0f

__device__ __forceinline__ v8f wmma_f16(v16h a, v16h b, v8f c) {
  // D = A(16x32 f16) * B(32x16 f16) + C(16x16 f32)
  return __builtin_amdgcn_wmma_f32_16x16x32_f16(
      /*neg_a=*/false, a, /*neg_b=*/false, b,
      /*c_mod=*/(short)0, c, /*reuse_a=*/false, /*reuse_b=*/false);
}

__device__ __forceinline__ v16h join8(v8h lo, v8h hi) {
  return __builtin_shufflevector(lo, hi, 0, 1, 2, 3, 4, 5, 6, 7,
                                         8, 9, 10, 11, 12, 13, 14, 15);
}

// Load a 16x32 f16 A-fragment from row-major LDS tile [16][KP].
// ISA layout: lanes 0-15 -> row M=lane, K = {koff..koff+7, koff+16..koff+23}
// with koff = kt*32 (+8 for lanes 16-31). Two 16-byte LDS loads per lane.
__device__ __forceinline__ v16h load_a_frag(const _Float16* base, int lane, int kt) {
  const int row  = lane & 15;
  const int koff = kt * 32 + ((lane & 16) ? 8 : 0);
  const _Float16* p = base + row * KP + koff;
  v8h lo = *(const v8h*)(p);        // K = koff .. koff+7
  v8h hi = *(const v8h*)(p + 16);   // K = koff+16 .. koff+23
  return join8(lo, hi);
}

// Load a lane-order pre-packed B-fragment (16 contiguous halfs per lane).
__device__ __forceinline__ v16h load_b_lds(const _Float16* p) {
  v8h lo = *(const v8h*)(p);
  v8h hi = *(const v8h*)(p + 8);
  return join8(lo, hi);
}

// Gather a 32x16 f16 B-fragment from a row-major f32 weight matrix W[rows][cols],
// zero-padding K rows >= rows and N cols >= cols. Done once at kernel start.
// ISA layout: lane n = lane&15; K = {kb..kb+7, kb+16..kb+23} (2 per VGPR),
// with kb = kt*32 (+8 for lanes 16-31), matching the A-fragment.
__device__ __forceinline__ v16h load_b_frag(const float* __restrict__ W,
                                            int rows, int cols,
                                            int kt, int ntile, int lane) {
  const int n  = ntile * 16 + (lane & 15);
  const int kb = kt * 32 + ((lane & 16) ? 8 : 0);
  v16h r;
#pragma unroll
  for (int v = 0; v < 8; ++v) {
    const int k0 = kb + 2 * v + ((v >= 4) ? 8 : 0);
    const float e0 = (k0     < rows && n < cols) ? W[(size_t)k0 * cols + n]       : 0.0f;
    const float e1 = (k0 + 1 < rows && n < cols) ? W[(size_t)(k0 + 1) * cols + n] : 0.0f;
    r[2 * v]     = (_Float16)e0;
    r[2 * v + 1] = (_Float16)e1;
  }
  return r;
}

__device__ __forceinline__ void store_b_lds(_Float16* dst, v16h f) {
  *(v8h*)(dst)     = __builtin_shufflevector(f, f, 0, 1, 2, 3, 4, 5, 6, 7);
  *(v8h*)(dst + 8) = __builtin_shufflevector(f, f, 8, 9, 10, 11, 12, 13, 14, 15);
}

__global__ __launch_bounds__(NTHREADS)
void snn_lif_fused(const float* __restrict__ x,
                   const float* __restrict__ Win1, const float* __restrict__ Wrec1,
                   const float* __restrict__ b1,   const float* __restrict__ leak1,
                   const float* __restrict__ Win2, const float* __restrict__ Wrec2,
                   const float* __restrict__ b2,   const float* __restrict__ leak2,
                   const float* __restrict__ Wout, const float* __restrict__ bout,
                   float* __restrict__ out) {
  // Ping-pong LDS tiles: x_t, layer-1 spikes, layer-2 spikes (f16, row-major 16x128).
  __shared__ _Float16 xs [2][MT * KP];
  __shared__ _Float16 s1s[2][MT * KP];
  __shared__ _Float16 s2s[2][MT * KP];
  // Lane-order packed B-fragments kept in LDS (saves VGPRs): 28 KB + 4 KB.
  __shared__ _Float16 wrec2f[4][NWAVES][32 * 16];
  __shared__ _Float16 woutf [4][32 * 16];

  const int tid  = threadIdx.x;
  const int lane = tid & 31;
  const int wid  = tid >> 5;          // wave id == N tile (neuron cols 16*wid..)
  const int b0   = blockIdx.x * MT;   // batch tile origin

  // ---- Pack hot weights into per-wave register B-fragments ----
  v16h fWin1[4], fWrec1[4], fWin2[4];
#pragma unroll
  for (int kt = 0; kt < 4; ++kt) {
    fWin1[kt]  = load_b_frag(Win1,  D_IN, H_N, kt, wid, lane);
    fWrec1[kt] = load_b_frag(Wrec1, H_N,  H_N, kt, wid, lane);
    fWin2[kt]  = load_b_frag(Win2,  H_N,  H_N, kt, wid, lane);
    // Cold weights -> LDS, packed in lane-fragment order.
    store_b_lds(&wrec2f[kt][wid][lane * 16],
                load_b_frag(Wrec2, H_N, H_N, kt, wid, lane));
    if (wid == 0)
      store_b_lds(&woutf[kt][lane * 16],
                  load_b_frag(Wout, H_N, O_N, kt, 0, lane));
  }

  // Per-lane neuron-column constants (C-fragment: N depends only on lane&15).
  const int   nloc = lane & 15;
  const int   ncol = wid * 16 + nloc;
  const float lb1  = (ncol < H_N) ? b1[ncol]    : 0.0f;
  const float lk1  = (ncol < H_N) ? leak1[ncol] : 0.0f;
  const float lb2  = (ncol < H_N) ? b2[ncol]    : 0.0f;
  const float lk2  = (ncol < H_N) ? leak2[ncol] : 0.0f;
  const float lbo  = (nloc < O_N) ? bout[nloc]  : 0.0f;
  const int   mhi  = (lane & 16) ? 8 : 0;       // C-fragment row offset

  // ---- Init LDS: zero spike buffers (pad cols stay 0 forever), load x_0 ----
  for (int i = tid; i < MT * KP; i += NTHREADS) {
    s1s[0][i] = (_Float16)0.0f; s1s[1][i] = (_Float16)0.0f;
    s2s[0][i] = (_Float16)0.0f; s2s[1][i] = (_Float16)0.0f;
  }
  for (int i = tid; i < (MT * KP) / 4; i += NTHREADS) {
    const int row = i >> 5;            // 32 float4 per 128-col row
    const int d4  = (i & 31) * 4;
    const v4f xv = *(const v4f*)&x[((size_t)(b0 + row) * T_LEN + 0) * D_IN + d4];
    *(v4h*)&xs[0][row * KP + d4] = __builtin_convertvector(xv, v4h);
  }
  __syncthreads();

  // Persistent membrane potentials (spikes derive from v: s = (v > 1)).
  float v1[8], v2[8];
#pragma unroll
  for (int r = 0; r < 8; ++r) { v1[r] = 0.0f; v2[r] = 0.0f; }

  for (int t = 0; t < T_LEN; ++t) {
    const int p = t & 1, q = p ^ 1;

    // Prefetch x_{t+1} into the other buffer (overlaps with phase-A WMMAs).
    if (t + 1 < T_LEN) {
      for (int i = tid; i < (MT * KP) / 4; i += NTHREADS) {
        const int row = i >> 5;
        const int d4  = (i & 31) * 4;
        const v4f xv =
            *(const v4f*)&x[((size_t)(b0 + row) * T_LEN + (t + 1)) * D_IN + d4];
        *(v4h*)&xs[q][row * KP + d4] = __builtin_convertvector(xv, v4h);
      }
    }

    // ---------------- Phase A: layer 1  i1 = x_t@Win1 + s1@Wrec1 + b1 ----------
    v8f accA, accB;
#pragma unroll
    for (int r = 0; r < 8; ++r) { accA[r] = lb1; accB[r] = 0.0f; }
#pragma unroll
    for (int kt = 0; kt < 4; ++kt)
      accA = wmma_f16(load_a_frag(xs[p],  lane, kt), fWin1[kt],  accA);
#pragma unroll
    for (int kt = 0; kt < 4; ++kt)
      accB = wmma_f16(load_a_frag(s1s[p], lane, kt), fWrec1[kt], accB);
#pragma unroll
    for (int r = 0; r < 8; ++r) {
      const float i1    = accA[r] + accB[r];
      const float sprev = (v1[r] > THRESH) ? 1.0f : 0.0f;
      v1[r] = lk1 * v1[r] * (1.0f - sprev) + i1;     // leak + reset-to-zero
      const float snew = (v1[r] > THRESH) ? 1.0f : 0.0f;
      s1s[q][(r + mhi) * KP + ncol] = (_Float16)snew;
    }

    // ---- Readout for step t-1 (wave 0), overlapped with phase A ----
    // s2s[p] holds s2 spikes of step t-1 (written as its q); the next write to
    // s2s[p] is phase B of step t+1, which is after bar1_{t+1} > bar1_t.
    if (wid == 0 && t > 0) {
      v8f oacc;
#pragma unroll
      for (int r = 0; r < 8; ++r) oacc[r] = lbo;
#pragma unroll
      for (int kt = 0; kt < 4; ++kt)
        oacc = wmma_f16(load_a_frag(s2s[p], lane, kt),
                        load_b_lds(&woutf[kt][lane * 16]), oacc);
      if (nloc < O_N) {
#pragma unroll
        for (int r = 0; r < 8; ++r)
          out[((size_t)(b0 + r + mhi) * T_LEN + (t - 1)) * O_N + nloc] = oacc[r];
      }
    }
    __syncthreads();

    // ---------------- Phase B: layer 2  i2 = s1_new@Win2 + s2@Wrec2 + b2 -------
#pragma unroll
    for (int r = 0; r < 8; ++r) { accA[r] = lb2; accB[r] = 0.0f; }
#pragma unroll
    for (int kt = 0; kt < 4; ++kt)
      accA = wmma_f16(load_a_frag(s1s[q], lane, kt), fWin2[kt], accA);
#pragma unroll
    for (int kt = 0; kt < 4; ++kt)
      accB = wmma_f16(load_a_frag(s2s[p], lane, kt),
                      load_b_lds(&wrec2f[kt][wid][lane * 16]), accB);
#pragma unroll
    for (int r = 0; r < 8; ++r) {
      const float i2    = accA[r] + accB[r];
      const float sprev = (v2[r] > THRESH) ? 1.0f : 0.0f;
      v2[r] = lk2 * v2[r] * (1.0f - sprev) + i2;
      const float snew = (v2[r] > THRESH) ? 1.0f : 0.0f;
      s2s[q][(r + mhi) * KP + ncol] = (_Float16)snew;
    }
    __syncthreads();
  }

  // ---- Epilogue: readout for the last step (t = T-1) ----
  // Last step's s2 spikes were written to s2s[q_last]; the final barrier in the
  // loop makes them visible.
  if (wid == 0) {
    const int qlast = (((T_LEN - 1) & 1) ^ 1);
    v8f oacc;
#pragma unroll
    for (int r = 0; r < 8; ++r) oacc[r] = lbo;
#pragma unroll
    for (int kt = 0; kt < 4; ++kt)
      oacc = wmma_f16(load_a_frag(s2s[qlast], lane, kt),
                      load_b_lds(&woutf[kt][lane * 16]), oacc);
    if (nloc < O_N) {
#pragma unroll
      for (int r = 0; r < 8; ++r)
        out[((size_t)(b0 + r + mhi) * T_LEN + (T_LEN - 1)) * O_N + nloc] = oacc[r];
    }
  }
}

extern "C" void kernel_launch(void* const* d_in, const int* in_sizes, int n_in,
                              void* d_out, int out_size, void* d_ws, size_t ws_size,
                              hipStream_t stream) {
  (void)in_sizes; (void)n_in; (void)out_size; (void)d_ws; (void)ws_size;
  const float* x     = (const float*)d_in[0];
  const float* Win1  = (const float*)d_in[1];
  const float* Wrec1 = (const float*)d_in[2];
  const float* b1    = (const float*)d_in[3];
  const float* leak1 = (const float*)d_in[4];
  const float* Win2  = (const float*)d_in[5];
  const float* Wrec2 = (const float*)d_in[6];
  const float* b2    = (const float*)d_in[7];
  const float* leak2 = (const float*)d_in[8];
  const float* Wout  = (const float*)d_in[9];
  const float* bout  = (const float*)d_in[10];
  float* out = (float*)d_out;

  dim3 grid(B_TOT / MT);     // 16 workgroups, one per 16-row batch tile
  dim3 block(NTHREADS);      // 7 wave32s: one per 16-neuron column tile
  snn_lif_fused<<<grid, block, 0, stream>>>(x, Win1, Wrec1, b1, leak1,
                                            Win2, Wrec2, b2, leak2,
                                            Wout, bout, out);
}
